// SGRU_69372311765505
// MI455X (gfx1250) — matrix operand order repaired and checked by reading, not compiled, not run on hardware
//
#include <hip/hip_runtime.h>
#include <math.h>

// Problem constants (match reference)
#define T_DIM 32
#define B_DIM 8
#define D_DIM 512
#define H_DIM 512
#define R_DIM 32
#define O_DIM 512
#define TWO_H (2 * H_DIM)
#define CLIP_VAL 1.0f
#define LN_EPS 1e-05f

typedef __attribute__((ext_vector_type(16))) __bf16 v16bf;
typedef __attribute__((ext_vector_type(8)))  float v8f;

// ---------------------------------------------------------------------------
// float4 helpers
// ---------------------------------------------------------------------------
__device__ __forceinline__ float4 ld4(const float* p) { return *(const float4*)p; }

__device__ __forceinline__ v16bf pack16(float4 q0, float4 q1, float4 q2, float4 q3) {
    v16bf r;
    r[0]  = (__bf16)q0.x; r[1]  = (__bf16)q0.y; r[2]  = (__bf16)q0.z; r[3]  = (__bf16)q0.w;
    r[4]  = (__bf16)q1.x; r[5]  = (__bf16)q1.y; r[6]  = (__bf16)q1.z; r[7]  = (__bf16)q1.w;
    r[8]  = (__bf16)q2.x; r[9]  = (__bf16)q2.y; r[10] = (__bf16)q2.z; r[11] = (__bf16)q2.w;
    r[12] = (__bf16)q3.x; r[13] = (__bf16)q3.y; r[14] = (__bf16)q3.z; r[15] = (__bf16)q3.w;
    return r;
}

// ---------------------------------------------------------------------------
// Vectorized copy (state init from inputs each call -> deterministic replay).
// n must be a multiple of 4 (true for all our buffers).
// ---------------------------------------------------------------------------
__global__ void copy_f32x4(const float* __restrict__ src, float* __restrict__ dst, size_t n4) {
    size_t i = (size_t)blockIdx.x * blockDim.x + threadIdx.x;
    if (i < n4) ((float4*)dst)[i] = ((const float4*)src)[i];
}

// ---------------------------------------------------------------------------
// WMMA GEMM:  C[M x N] = A[M x K] @ W[N x K]^T + bias[N]
// bf16 operands, f32 accumulate via v_wmma_f32_16x16x32_bf16 (wave32).
// One wave per 16x16 tile; K unrolled x2 with independent accumulators.
// Fragment packing follows ISA 7.12.2 16-bit layouts; all fragment data is
// fetched as contiguous float4 (B128) runs:
//   A (16x32): lane half 'hi' -> K = k0+8*hi+{0..7} and k0+16+8*hi+{0..7}
//   B (32x16): lane half 'hi' -> K = k0+8*hi+{0..7} and k0+16+8*hi+{0..7}
// Out-of-range A rows are clamped to row 0 (their C rows are never stored).
// ---------------------------------------------------------------------------
__global__ void wmma_gemm_wT(const float* __restrict__ A, int M, int K,
                             const float* __restrict__ W,
                             const float* __restrict__ bias,
                             float* __restrict__ C, int ldc, int Ntiles) {
    const int tile = blockIdx.x;
    const int mt = tile / Ntiles;
    const int nt = tile % Ntiles;
    const int lane = threadIdx.x & 31;
    const int lo16 = lane & 15;
    const int hi = lane >> 4;

    int m = mt * 16 + lo16;
    if (m >= M) m = 0;                       // clamp: padded rows discarded at store
    const int n = nt * 16 + lo16;
    const float* __restrict__ arow = A + (size_t)m * K + 8 * hi;
    const float* __restrict__ wrow = W + (size_t)n * K + 8 * hi;

    v8f acc0 = {}, acc1 = {};
    for (int k0 = 0; k0 < K; k0 += 64) {
        // ---- k-step 0: K slice [k0, k0+32) ----
        {
            v16bf af = pack16(ld4(arow + k0),      ld4(arow + k0 + 4),
                              ld4(arow + k0 + 16), ld4(arow + k0 + 20));
            v16bf bf = pack16(ld4(wrow + k0),      ld4(wrow + k0 + 4),
                              ld4(wrow + k0 + 16), ld4(wrow + k0 + 20));
            acc0 = __builtin_amdgcn_wmma_f32_16x16x32_bf16(
                false, af, false, bf, (short)0, acc0, false, false);
        }
        // ---- k-step 1: K slice [k0+32, k0+64) (independent accumulator) ----
        {
            const int k1 = k0 + 32;
            v16bf af = pack16(ld4(arow + k1),      ld4(arow + k1 + 4),
                              ld4(arow + k1 + 16), ld4(arow + k1 + 20));
            v16bf bf = pack16(ld4(wrow + k1),      ld4(wrow + k1 + 4),
                              ld4(wrow + k1 + 16), ld4(wrow + k1 + 20));
            acc1 = __builtin_amdgcn_wmma_f32_16x16x32_bf16(
                false, af, false, bf, (short)0, acc1, false, false);
        }
    }

    const float bv = bias[n];                // bias always present; hoisted load
#pragma unroll
    for (int r = 0; r < 8; ++r) {
        const int mr = mt * 16 + r + 8 * hi;
        if (mr < M) {
            C[mr * ldc + n] = acc0[r] + acc1[r] + bv;   // 32-bit index math
        }
    }
}

// ---------------------------------------------------------------------------
// plast[b,i] = softplus(alpha) * sum_j dU[b,i,j] * h[b,j]
// One wave per output element; float4 (B128) streaming; shuffle reduction.
// ---------------------------------------------------------------------------
__global__ void plast_kernel(const float* __restrict__ dU, const float* __restrict__ h,
                             const float* __restrict__ alpha_p, float* __restrict__ plast) {
    const int wave = blockIdx.x * (blockDim.x >> 5) + (threadIdx.x >> 5);
    const int lane = threadIdx.x & 31;
    const int b = wave / H_DIM;
    const int i = wave % H_DIM;
    const float* __restrict__ row = dU + ((size_t)b * H_DIM + i) * H_DIM;
    const float* __restrict__ hv  = h + (size_t)b * H_DIM;
    float s = 0.0f;
#pragma unroll
    for (int it = 0; it < H_DIM / 128; ++it) {          // 4 iters of float4 per lane
        const int j = it * 128 + lane * 4;
        const float4 a = ld4(row + j);
        const float4 c = ld4(hv + j);
        s = fmaf(a.x, c.x, s); s = fmaf(a.y, c.y, s);
        s = fmaf(a.z, c.z, s); s = fmaf(a.w, c.w, s);
    }
#pragma unroll
    for (int off = 16; off > 0; off >>= 1) s += __shfl_down(s, off, 32);
    if (lane == 0) {
        const float a = alpha_p[0];
        const float spa = (a > 20.0f) ? a : log1pf(expf(a));
        plast[wave] = spa * s;
    }
}

// ---------------------------------------------------------------------------
// Fused: Wx = LN(Wxr), Wh = LN(Whr + plast on upper half), gate, v/h update.
// One block (256 thr) per batch row; LDS reductions over 2H = 1024.
// Wxr/Whr already include the linear biases (added in GEMM epilogue).
// ---------------------------------------------------------------------------
__global__ void ln_gate_kernel(const float* __restrict__ Wxr, const float* __restrict__ Whr,
                               const float* __restrict__ plast,
                               const float* __restrict__ lnx_g, const float* __restrict__ lnx_b,
                               const float* __restrict__ lnh_g, const float* __restrict__ lnh_b,
                               float* __restrict__ v, float* __restrict__ h_nxt,
                               float* __restrict__ hs_t) {
    const int b = blockIdx.x;
    const int tid = threadIdx.x;          // 256 threads
    __shared__ float red[256];

    float wx[4], wh[4];
    float sx = 0.f, sx2 = 0.f, sh = 0.f, sh2 = 0.f;
#pragma unroll
    for (int q = 0; q < 4; ++q) {
        const int j = tid + q * 256;      // 0..1023
        float a = Wxr[b * TWO_H + j];
        float c = Whr[b * TWO_H + j];
        if (j >= H_DIM) c += plast[b * H_DIM + (j - H_DIM)];
        wx[q] = a; wh[q] = c;
        sx += a; sx2 += a * a; sh += c; sh2 += c * c;
    }

    auto blockSum = [&](float val) -> float {
        red[tid] = val; __syncthreads();
        for (int s = 128; s > 0; s >>= 1) {
            if (tid < s) red[tid] += red[tid + s];
            __syncthreads();
        }
        float r = red[0]; __syncthreads();
        return r;
    };

    const float inv2h = 1.0f / (float)TWO_H;
    const float mx  = blockSum(sx)  * inv2h;
    const float m2x = blockSum(sx2) * inv2h;
    const float mh  = blockSum(sh)  * inv2h;
    const float m2h = blockSum(sh2) * inv2h;
    const float rsx = rsqrtf(fmaxf(m2x - mx * mx, 0.0f) + LN_EPS);
    const float rsh = rsqrtf(fmaxf(m2h - mh * mh, 0.0f) + LN_EPS);

#pragma unroll
    for (int q = 0; q < 2; ++q) {
        const int i  = tid + q * 256;     // 0..511
        const int jd = i + H_DIM;
        const float yz = lnx_g[i]  * (wx[q]     - mx) * rsx + lnx_b[i]
                       + lnh_g[i]  * (wh[q]     - mh) * rsh + lnh_b[i];
        const float yd = lnx_g[jd] * (wx[2 + q] - mx) * rsx + lnx_b[jd]
                       + lnh_g[jd] * (wh[2 + q] - mh) * rsh + lnh_b[jd];
        const float z  = 1.0f / (1.0f + expf(-yz));
        const int o = b * H_DIM + i;
        const float vn = (1.0f - z) * v[o] + z * yd;
        const float nh = fmaxf(vn, 0.0f);
        v[o] = vn;
        h_nxt[o] = nh;
        hs_t[o] = nh;
    }
}

// ---------------------------------------------------------------------------
// mod MLP: hid = relu(nh @ h2mod_w^T + b)  (R=32), mod = hid @ mod2h_w^T + b (3)
// One block per batch row. Outputs r=sig(mod0), s=sig(mod1), m=mod2.
// ---------------------------------------------------------------------------
__global__ void mod_kernel(const float* __restrict__ h_nxt,
                           const float* __restrict__ h2mod_w, const float* __restrict__ h2mod_b,
                           const float* __restrict__ mod2h_w, const float* __restrict__ mod2h_b,
                           float* __restrict__ Rb, float* __restrict__ Sb, float* __restrict__ Mb) {
    const int b = blockIdx.x;
    const int tid = threadIdx.x;                // 256
    __shared__ float part[256];
    __shared__ float hid[R_DIM];

    const int o = tid & 31;                     // output index
    const int chunk = tid >> 5;                 // 8 chunks of 64 K each
    const float* __restrict__ nh = h_nxt + b * H_DIM;
    const float* __restrict__ wr = h2mod_w + o * H_DIM;
    float s = 0.0f;
    const int k0 = chunk * 64;
#pragma unroll
    for (int k = 0; k < 64; k += 4) {
        const float4 a = ld4(nh + k0 + k);
        const float4 w = ld4(wr + k0 + k);
        s = fmaf(a.x, w.x, s); s = fmaf(a.y, w.y, s);
        s = fmaf(a.z, w.z, s); s = fmaf(a.w, w.w, s);
    }
    part[tid] = s;
    __syncthreads();
    if (tid < R_DIM) {
        float acc = 0.0f;
#pragma unroll
        for (int c = 0; c < 8; ++c) acc += part[c * 32 + tid];
        hid[tid] = fmaxf(acc + h2mod_b[tid], 0.0f);
    }
    __syncthreads();
    if (tid < 3) {
        float acc = 0.0f;
#pragma unroll
        for (int oo = 0; oo < R_DIM; ++oo) acc = fmaf(mod2h_w[tid * R_DIM + oo], hid[oo], acc);
        acc += mod2h_b[tid];
        if (tid == 0) Rb[b] = 1.0f / (1.0f + expf(-acc));
        if (tid == 1) Sb[b] = 1.0f / (1.0f + expf(-acc));
        if (tid == 2) Mb[b] = acc;
    }
}

// ---------------------------------------------------------------------------
// te[b,j] = (1-r[b])*te[b,j] + r[b]*h_prev[b,j]     (in place, float4)
// ---------------------------------------------------------------------------
__global__ void te_update_kernel(const float* __restrict__ h_cur,
                                 const float* __restrict__ Rb, float* __restrict__ te) {
    const int i4 = blockIdx.x * blockDim.x + threadIdx.x;   // over (B*H)/4
    if (i4 >= (B_DIM * H_DIM) / 4) return;
    const int b = (i4 * 4) / H_DIM;
    const float r = Rb[b];
    const float4 t = ((const float4*)te)[i4];
    const float4 h = ((const float4*)h_cur)[i4];
    float4 o;
    o.x = (1.0f - r) * t.x + r * h.x;
    o.y = (1.0f - r) * t.y + r * h.y;
    o.z = (1.0f - r) * t.z + r * h.z;
    o.w = (1.0f - r) * t.w + r * h.w;
    ((float4*)te)[i4] = o;
}

// ---------------------------------------------------------------------------
// Trace update over (B,H,H): tE, dU fused read-modify-write (L2 resident).
// float4 per thread (4 consecutive j). Clip bounds recomputed inline from the
// lower block of h2h_w (L2 hit, avoids a precompute pass + extra state).
// ---------------------------------------------------------------------------
__global__ void trace_update_kernel(const float* __restrict__ h_nxt, const float* __restrict__ te,
                                    const float* __restrict__ Sb, const float* __restrict__ Mb,
                                    float* __restrict__ tE, float* __restrict__ dU,
                                    const float* __restrict__ h2h_w,
                                    const float* __restrict__ alpha_p,
                                    const float* __restrict__ tau_p) {
    const size_t i4 = (size_t)blockIdx.x * blockDim.x + threadIdx.x;  // over B*H*H/4
    if (i4 >= (size_t)B_DIM * H_DIM * H_DIM / 4) return;
    const size_t idx = i4 * 4;
    const int j = (int)(idx % H_DIM);           // multiple of 4
    const size_t t2 = idx / H_DIM;
    const int i = (int)(t2 % H_DIM);
    const int b = (int)(t2 / H_DIM);

    const float s   = Sb[b];
    const float m   = Mb[b];
    const float a   = alpha_p[0];
    const float spa = (a > 20.0f) ? a : log1pf(expf(a));
    const float tau = 1.0f / (1.0f + expf(-tau_p[0]));
    const float inv = 1.0f / (spa + 1e-8f);
    const float tmm = tau * m;

    const float* __restrict__ nh  = h_nxt + b * H_DIM;
    const float* __restrict__ tev = te    + b * H_DIM;

    const float nhi = nh[i];
    const float tei = tev[i];
    const float4 nhj = ld4(nh + j);
    const float4 tej = ld4(tev + j);
    const float4 tE4 = ((const float4*)tE)[i4];
    const float4 dU4 = ((const float4*)dU)[i4];
    const float4 wdv = ld4(h2h_w + (size_t)(H_DIM + i) * H_DIM + j);

    float4 tEn, dUn;
#define TRACE_LANE(c)                                                          \
    {                                                                          \
        const float skew = nhi * tej.c - nhj.c * tei;                          \
        const float te_n = (1.0f - s) * tE4.c + s * skew;                      \
        const float up =  fmaxf(CLIP_VAL - wdv.c, 0.0f) * inv;                 \
        const float lo = -fmaxf(CLIP_VAL + wdv.c, 0.0f) * inv;                 \
        float du = (1.0f - tau) * dU4.c + tmm * te_n;                          \
        du = (du > up) ? up : du;                                              \
        du = (du < lo) ? lo : du;                                              \
        tEn.c = te_n; dUn.c = du;                                              \
    }
    TRACE_LANE(x) TRACE_LANE(y) TRACE_LANE(z) TRACE_LANE(w)
#undef TRACE_LANE

    ((float4*)tE)[i4] = tEn;
    ((float4*)dU)[i4] = dUn;
}

// ---------------------------------------------------------------------------
// Host-side orchestration (graph-captured; all launches on `stream`)
// ---------------------------------------------------------------------------
extern "C" void kernel_launch(void* const* d_in, const int* in_sizes, int n_in,
                              void* d_out, int out_size, void* d_ws, size_t ws_size,
                              hipStream_t stream) {
    (void)in_sizes; (void)n_in; (void)out_size; (void)ws_size;

    const float* x        = (const float*)d_in[0];   // (T,B,D)
    const float* h0       = (const float*)d_in[1];   // (B,H)
    const float* v0       = (const float*)d_in[2];
    const float* dU0      = (const float*)d_in[3];   // (B,H,H)
    const float* te0      = (const float*)d_in[4];
    const float* tE0      = (const float*)d_in[5];   // (B,H,H)
    const float* x2h_w    = (const float*)d_in[6];   // (2H,D)
    const float* x2h_b    = (const float*)d_in[7];
    const float* h2h_w    = (const float*)d_in[8];   // (2H,H)
    const float* h2h_b    = (const float*)d_in[9];
    const float* lnx_g    = (const float*)d_in[10];
    const float* lnx_b    = (const float*)d_in[11];
    const float* lnh_g    = (const float*)d_in[12];
    const float* lnh_b    = (const float*)d_in[13];
    const float* h2mod_w  = (const float*)d_in[14];  // (R,H)
    const float* h2mod_b  = (const float*)d_in[15];
    const float* mod2h_w  = (const float*)d_in[16];  // (3,R)
    const float* mod2h_b  = (const float*)d_in[17];
    const float* alpha    = (const float*)d_in[18];
    const float* tau_U    = (const float*)d_in[19];
    const float* dec_w    = (const float*)d_in[20];  // (O,H)
    const float* dec_b    = (const float*)d_in[21];

    float* out = (float*)d_out;

    // Workspace layout (floats); every offset is a multiple of 16 floats
    float* ws = (float*)d_ws;
    size_t off = 0;
    const size_t BH  = (size_t)B_DIM * H_DIM;          // 4096
    const size_t BHH = (size_t)B_DIM * H_DIM * H_DIM;  // 2 097 152
    float* hA    = ws + off; off += BH;
    float* hB    = ws + off; off += BH;
    float* vbuf  = ws + off; off += BH;
    float* tebuf = ws + off; off += BH;
    float* plast = ws + off; off += BH;
    float* Wxr   = ws + off; off += (size_t)B_DIM * TWO_H;  // 8192
    float* Whr   = ws + off; off += (size_t)B_DIM * TWO_H;
    float* Rb    = ws + off; off += 16;
    float* Sb    = ws + off; off += 16;
    float* Mb    = ws + off; off += 16;
    float* hs    = ws + off; off += (size_t)T_DIM * BH;     // 131072
    float* dU    = ws + off; off += BHH;
    float* tE    = ws + off; off += BHH;

    // --- state init from inputs (deterministic each call) ---
    copy_f32x4<<<(unsigned)((BH / 4  + 255) / 256), 256, 0, stream>>>(h0,  hA,    BH / 4);
    copy_f32x4<<<(unsigned)((BH / 4  + 255) / 256), 256, 0, stream>>>(v0,  vbuf,  BH / 4);
    copy_f32x4<<<(unsigned)((BH / 4  + 255) / 256), 256, 0, stream>>>(te0, tebuf, BH / 4);
    copy_f32x4<<<(unsigned)((BHH / 4 + 255) / 256), 256, 0, stream>>>(dU0, dU,    BHH / 4);
    copy_f32x4<<<(unsigned)((BHH / 4 + 255) / 256), 256, 0, stream>>>(tE0, tE,    BHH / 4);

    const int NT_2H  = TWO_H / 16;            // 64 N tiles for x2h / h2h
    const int NT_O   = O_DIM / 16;            // 32 N tiles for decoder
    const int MT_DEC = (T_DIM * B_DIM) / 16;  // 16 M tiles for decoder

    for (int t = 0; t < T_DIM; ++t) {
        const float* hc = (t & 1) ? hB : hA;   // previous hidden
        float*       hn = (t & 1) ? hA : hB;   // next hidden

        // 1) plast = softplus(alpha) * (dU . h_prev)
        plast_kernel<<<(unsigned)(BH / 8), 256, 0, stream>>>(dU, hc, alpha, plast);

        // 2) Wxr = x_t @ x2h_w^T + x2h_b ; Whr = h @ h2h_w^T + h2h_b  (bf16 WMMA)
        wmma_gemm_wT<<<NT_2H, 32, 0, stream>>>(x + (size_t)t * B_DIM * D_DIM,
                                               B_DIM, D_DIM, x2h_w, x2h_b, Wxr, TWO_H, NT_2H);
        wmma_gemm_wT<<<NT_2H, 32, 0, stream>>>(hc, B_DIM, H_DIM, h2h_w, h2h_b, Whr, TWO_H, NT_2H);

        // 3) dual LayerNorm + gate + ReLU -> v, h_next, hs[t]
        ln_gate_kernel<<<B_DIM, 256, 0, stream>>>(Wxr, Whr, plast,
                                                  lnx_g, lnx_b, lnh_g, lnh_b,
                                                  vbuf, hn, hs + (size_t)t * BH);

        // 4) mod MLP -> r, s, m per batch
        mod_kernel<<<B_DIM, 256, 0, stream>>>(hn, h2mod_w, h2mod_b, mod2h_w, mod2h_b,
                                              Rb, Sb, Mb);

        // 5) te update (needs h_prev)
        te_update_kernel<<<(unsigned)((BH / 4 + 255) / 256), 256, 0, stream>>>(hc, Rb, tebuf);

        // 6) fused tE/dU trace update + clip (L2-resident streaming, B128)
        trace_update_kernel<<<(unsigned)(BHH / 4 / 256), 256, 0, stream>>>(hn, tebuf, Sb, Mb,
                                                                           tE, dU, h2h_w,
                                                                           alpha, tau_U);
    }

    // Decoder: out = hs @ dec_w^T + dec_b   (256 x 512 x 512, bf16 WMMA)
    wmma_gemm_wT<<<MT_DEC * NT_O, 32, 0, stream>>>(hs, T_DIM * B_DIM, H_DIM,
                                                   dec_w, dec_b, out, O_DIM, NT_O);
}
